// GraphAttentionLayer_29283087024215
// MI455X (gfx1250) — matrix operand order repaired and checked by reading, not compiled, not run on hardware
//
#include <hip/hip_runtime.h>
#include <hip/hip_bf16.h>

#define N_NODES 8192
#define F_IN    512
#define F_OUT   256
#define ALPHA   0.2f

typedef __bf16 bf16;
typedef __attribute__((ext_vector_type(16))) bf16  bf16x16;
typedef __attribute__((ext_vector_type(8)))  bf16  bf16x8;
typedef __attribute__((ext_vector_type(8)))  float f32x8;
typedef __attribute__((ext_vector_type(8)))  float v8f;
typedef __attribute__((ext_vector_type(4)))  float f32x4;
typedef __attribute__((ext_vector_type(4)))  int   i32x4;
typedef __attribute__((ext_vector_type(8)))  int   i32x8;
typedef __attribute__((ext_vector_type(4)))  unsigned int u32x4;

// ---------------------------------------------------------------------------
// TDM: issue an async load of a 32x64 int32 tile of adj into LDS.
// D# built per CDNA5 ISA §8.3-8.6: 2D tensor, data_size=4B, tile 64(x) x 32(y),
// row stride 8192 elements, descriptor type=2.
// ---------------------------------------------------------------------------
__device__ __forceinline__ void tdm_load_adj_tile(const int* tile_gaddr,
                                                  unsigned int lds_byte_off) {
    unsigned long long ga = (unsigned long long)(uintptr_t)tile_gaddr;
    u32x4 g0;
    g0[0] = 1u;                                            // count=1, user mode
    g0[1] = lds_byte_off;                                  // lds_addr
    g0[2] = (unsigned int)(ga & 0xFFFFFFFFu);              // global_addr[31:0]
    g0[3] = (unsigned int)((ga >> 32) & 0x01FFFFFFu)       // global_addr[56:32]
          | (2u << 30);                                    // type=2 (image)
    i32x8 g1;
    g1[0] = (int)(2u << 16);                               // data_size=2 -> 4 bytes
    g1[1] = (int)(((unsigned)N_NODES & 0xFFFFu) << 16);    // tensor_dim0[15:0]
    g1[2] = (int)((((unsigned)N_NODES >> 16) & 0xFFFFu)    // tensor_dim0[31:16]
          | (((unsigned)N_NODES & 0xFFFFu) << 16));        // tensor_dim1[15:0]
    g1[3] = (int)((((unsigned)N_NODES >> 16) & 0xFFFFu)    // tensor_dim1[31:16]
          | (64u << 16));                                  // tile_dim0 = 64
    g1[4] = 32;                                            // tile_dim1 = 32
    g1[5] = N_NODES;                                       // tensor_dim0_stride[31:0]
    g1[6] = 0;                                             // stride hi / dim1_stride lo
    g1[7] = 0;
    i32x4 g2 = {0, 0, 0, 0};
    i32x4 g3 = {0, 0, 0, 0};
#if __clang_major__ >= 23
    i32x8 g4 = {0, 0, 0, 0, 0, 0, 0, 0};
    __builtin_amdgcn_tensor_load_to_lds(g0, g1, g2, g3, g4, 0);
#else
    __builtin_amdgcn_tensor_load_to_lds(g0, g1, g2, g3, 0);
#endif
}

// ---------------------------------------------------------------------------
// Kernel 1: WT[n][k] = bf16(W[k][n])   (B operand needs N-major, K-contiguous)
// ---------------------------------------------------------------------------
__global__ void wt_convert_kernel(const float* __restrict__ W, bf16* __restrict__ WT) {
    int idx = blockIdx.x * 256 + threadIdx.x;   // 512*256 = 131072 total
    int k = idx >> 8;
    int n = idx & 255;
    WT[n * F_IN + k] = (bf16)W[k * F_OUT + n];
}

// ---------------------------------------------------------------------------
// Kernel 2: h = x @ W via bf16 WMMA; store transposed hT[F_OUT][N] in bf16.
// One wave per 16x16 output tile. 8 waves / block.
// ---------------------------------------------------------------------------
__global__ void gemm1_kernel(const float* __restrict__ x, const bf16* __restrict__ WT,
                             bf16* __restrict__ hT) {
    int wave = threadIdx.x >> 5;
    int lane = threadIdx.x & 31;
    int tile = blockIdx.x * 8 + wave;          // 512 * 16 = 8192 tiles
    int mtile = tile >> 4;
    int ntile = tile & 15;
    int hi = lane >> 4;
    int lo = lane & 15;

    const float* xrow = x  + (size_t)(mtile * 16 + lo) * F_IN;
    const bf16*  wrow = WT + (size_t)(ntile * 16 + lo) * F_IN;

    v8f acc = {};
    for (int k0 = 0; k0 < F_IN; k0 += 32) {
        f32x8 a0 = *(const f32x8*)(xrow + k0 + hi * 16);
        f32x8 a1 = *(const f32x8*)(xrow + k0 + hi * 16 + 8);
        bf16x16 A;
        #pragma unroll
        for (int t = 0; t < 8; ++t) { A[t] = (bf16)a0[t]; A[t + 8] = (bf16)a1[t]; }
        bf16x16 B = *(const bf16x16*)(wrow + k0 + hi * 16);
        acc = __builtin_amdgcn_wmma_f32_16x16x32_bf16(false, A, false, B,
                                                      (short)0, acc, false, false);
    }
    bf16x8 o;
    #pragma unroll
    for (int v = 0; v < 8; ++v) o[v] = (bf16)acc[v];
    *(bf16x8*)(hT + (size_t)(ntile * 16 + lo) * N_NODES + mtile * 16 + 8 * hi) = o;
}

// ---------------------------------------------------------------------------
// Kernel 3: f1[i] = h[i,:].a1 ; f2[i] = h[i,:].a2
// ---------------------------------------------------------------------------
__global__ void f1f2_kernel(const bf16* __restrict__ hT, const float* __restrict__ a,
                            float* __restrict__ f1, float* __restrict__ f2) {
    int i = blockIdx.x * 256 + threadIdx.x;    // 8192
    float s1 = 0.f, s2 = 0.f;
    for (int c = 0; c < F_OUT; ++c) {
        float hv = (float)hT[(size_t)c * N_NODES + i];
        s1 += hv * a[c];
        s2 += hv * a[F_OUT + c];
    }
    f1[i] = s1;
    f2[i] = s2;
}

// ---------------------------------------------------------------------------
// Kernel 4: per-row online softmax stats (max, 1/sum). Wave per row.
// ---------------------------------------------------------------------------
__global__ void rowstats_kernel(const int* __restrict__ adj, const float* __restrict__ f1,
                                const float* __restrict__ f2, float* __restrict__ mrow,
                                float* __restrict__ sinv) {
    int wave = threadIdx.x >> 5;
    int lane = threadIdx.x & 31;
    int row  = blockIdx.x * 8 + wave;
    float fr = f1[row];
    const int* arow = adj + (size_t)row * N_NODES;

    float m = -1e30f, s = 0.f;
    for (int j0 = 0; j0 < N_NODES; j0 += 128) {
        __builtin_prefetch(arow + j0 + 1024, 0, 1);
        i32x4 av = *(const i32x4*)(arow + j0 + lane * 4);
        f32x4 fv = *(const f32x4*)(f2 + j0 + lane * 4);
        #pragma unroll
        for (int e = 0; e < 4; ++e) {
            if (av[e] > 0) {
                float sc = fr + fv[e];
                sc = sc > 0.f ? sc : ALPHA * sc;
                float nm = fmaxf(m, sc);
                s = s * __expf(m - nm) + __expf(sc - nm);
                m = nm;
            }
        }
    }
    #pragma unroll
    for (int off = 16; off >= 1; off >>= 1) {
        float m2 = __shfl_xor(m, off, 32);
        float s2 = __shfl_xor(s, off, 32);
        float nm = fmaxf(m, m2);
        s = s * __expf(m - nm) + s2 * __expf(m2 - nm);
        m = nm;
    }
    if (lane == 0) {
        mrow[row] = m;
        sinv[row] = (s > 0.f) ? 1.f / s : 0.f;
    }
}

// ---------------------------------------------------------------------------
// Kernel 5: fused  out = elu( softmax(mask(e)) @ h ).
// Block = 32 output rows, 256 threads = 8 waves; wave w owns output features
// [32w, 32w+32) and both 16-row M-tiles. adj tiles are TDM double-buffered in
// LDS; attention tile P[32][64] (bf16, exp/normalized folded in) regenerated
// per K-chunk and consumed by 8 WMMAs per wave per chunk.
// ---------------------------------------------------------------------------
__global__ void attn_kernel(const int* __restrict__ adj, const bf16* __restrict__ hT,
                            const float* __restrict__ f1, const float* __restrict__ f2,
                            const float* __restrict__ mrow, const float* __restrict__ sinv,
                            float* __restrict__ out) {
    __shared__ int  adjbuf[2][32 * 64];   // 16 KB, TDM double buffer
    __shared__ bf16 P[32][64];            // 4 KB attention tile

    int tid  = threadIdx.x;
    int wave = tid >> 5;
    int lane = tid & 31;
    int hi   = lane >> 4;
    int lo   = lane & 15;
    int i0   = blockIdx.x * 32;

    // producer role: 8 threads per row, 8 cols each
    int pr = tid >> 3;          // 0..31
    int pc = (tid & 7) * 8;     // 0..56
    int prow = i0 + pr;
    float fr = f1[prow];
    float mr = mrow[prow];
    float si = sinv[prow];

    const int* atile0 = adj + (size_t)i0 * N_NODES;   // block's 32-row band, col 0

    // prologue: wave 0 kicks off TDM for chunk 0
    if (wave == 0) {
        tdm_load_adj_tile(atile0, (unsigned int)(uintptr_t)&adjbuf[0][0]);
    }

    v8f acc00 = {}, acc01 = {}, acc10 = {}, acc11 = {};

    for (int jc = 0; jc < N_NODES / 64; ++jc) {
        int cur = jc & 1;
        int j0  = jc * 64;
        if (wave == 0) {
            __builtin_amdgcn_s_wait_tensorcnt(0);      // adjbuf[cur] landed
        }
        __syncthreads();  // adj tile visible; prev iter's P reads complete
        if (wave == 0 && jc + 1 < N_NODES / 64) {
            tdm_load_adj_tile(atile0 + (j0 + 64),
                              (unsigned int)(uintptr_t)&adjbuf[1 - cur][0]);
        }

        // regenerate P tile: P = adj ? exp(lrelu(f1+f2) - m) / s : 0   (bf16)
        {
            const int* arow = &adjbuf[cur][pr * 64];
            i32x4 av0 = *(const i32x4*)(arow + pc);
            i32x4 av1 = *(const i32x4*)(arow + pc + 4);
            f32x8 fv  = *(const f32x8*)(f2 + j0 + pc);
            bf16x8 pv;
            #pragma unroll
            for (int e = 0; e < 4; ++e) {
                float sc0 = fr + fv[e];
                sc0 = sc0 > 0.f ? sc0 : ALPHA * sc0;
                pv[e] = (av0[e] > 0) ? (bf16)(__expf(sc0 - mr) * si) : (bf16)0.f;
                float sc1 = fr + fv[e + 4];
                sc1 = sc1 > 0.f ? sc1 : ALPHA * sc1;
                pv[e + 4] = (av1[e] > 0) ? (bf16)(__expf(sc1 - mr) * si) : (bf16)0.f;
            }
            *(bf16x8*)(&P[pr][pc]) = pv;
        }
        __syncthreads();  // P visible to all waves

        #pragma unroll
        for (int kt = 0; kt < 2; ++kt) {
            int kk = kt * 32;
            // A fragments (16-bit A layout, contiguous k = 16*hi + s), two M-tiles
            bf16x16 A0, A1;
            {
                bf16x8 a0lo = *(const bf16x8*)(&P[lo][kk + hi * 16]);
                bf16x8 a0hi = *(const bf16x8*)(&P[lo][kk + hi * 16 + 8]);
                bf16x8 a1lo = *(const bf16x8*)(&P[16 + lo][kk + hi * 16]);
                bf16x8 a1hi = *(const bf16x8*)(&P[16 + lo][kk + hi * 16 + 8]);
                #pragma unroll
                for (int t = 0; t < 8; ++t) {
                    A0[t] = a0lo[t]; A0[t + 8] = a0hi[t];
                    A1[t] = a1lo[t]; A1[t + 8] = a1hi[t];
                }
            }
            // B fragments: hT is N-major so each lane's 16 K-values are contiguous
            bf16x16 B0 = *(const bf16x16*)(hT + (size_t)(wave * 32 + lo)      * N_NODES + j0 + kk + hi * 16);
            bf16x16 B1 = *(const bf16x16*)(hT + (size_t)(wave * 32 + 16 + lo) * N_NODES + j0 + kk + hi * 16);
            acc00 = __builtin_amdgcn_wmma_f32_16x16x32_bf16(false, A0, false, B0, (short)0, acc00, false, false);
            acc01 = __builtin_amdgcn_wmma_f32_16x16x32_bf16(false, A0, false, B1, (short)0, acc01, false, false);
            acc10 = __builtin_amdgcn_wmma_f32_16x16x32_bf16(false, A1, false, B0, (short)0, acc10, false, false);
            acc11 = __builtin_amdgcn_wmma_f32_16x16x32_bf16(false, A1, false, B1, (short)0, acc11, false, false);
        }
    }

    // epilogue: ELU, scatter D layout (M = v + 8*hi, N = lo)
    #pragma unroll
    for (int v = 0; v < 8; ++v) {
        int M0 = i0 + v + 8 * hi;
        int M1 = M0 + 16;
        int N0 = wave * 32 + lo;
        float e00 = acc00[v]; e00 = e00 > 0.f ? e00 : __expf(e00) - 1.f;
        float e01 = acc01[v]; e01 = e01 > 0.f ? e01 : __expf(e01) - 1.f;
        float e10 = acc10[v]; e10 = e10 > 0.f ? e10 : __expf(e10) - 1.f;
        float e11 = acc11[v]; e11 = e11 > 0.f ? e11 : __expf(e11) - 1.f;
        out[(size_t)M0 * F_OUT + N0]      = e00;
        out[(size_t)M0 * F_OUT + N0 + 16] = e01;
        out[(size_t)M1 * F_OUT + N0]      = e10;
        out[(size_t)M1 * F_OUT + N0 + 16] = e11;
    }
}

// ---------------------------------------------------------------------------
extern "C" void kernel_launch(void* const* d_in, const int* in_sizes, int n_in,
                              void* d_out, int out_size, void* d_ws, size_t ws_size,
                              hipStream_t stream) {
    (void)in_sizes; (void)n_in; (void)out_size; (void)ws_size;
    const float* x   = (const float*)d_in[0];
    const int*   adj = (const int*)d_in[1];
    const float* W   = (const float*)d_in[2];
    const float* a   = (const float*)d_in[3];
    float*       out = (float*)d_out;

    char* ws = (char*)d_ws;
    bf16*  hT   = (bf16*)ws;                                               // 4 MB
    bf16*  WT   = (bf16*)(ws + (size_t)F_OUT * N_NODES * sizeof(bf16));    // 256 KB
    float* f1   = (float*)(ws + (size_t)F_OUT * N_NODES * sizeof(bf16)
                              + (size_t)F_OUT * F_IN    * sizeof(bf16));
    float* f2   = f1 + N_NODES;
    float* mr   = f2 + N_NODES;
    float* si   = mr + N_NODES;

    hipLaunchKernelGGL(wt_convert_kernel, dim3(512),  dim3(256), 0, stream, W, WT);
    hipLaunchKernelGGL(gemm1_kernel,      dim3(1024), dim3(256), 0, stream, x, WT, hT);
    hipLaunchKernelGGL(f1f2_kernel,       dim3(32),   dim3(256), 0, stream, hT, a, f1, f2);
    hipLaunchKernelGGL(rowstats_kernel,   dim3(1024), dim3(256), 0, stream, adj, f1, f2, mr, si);
    hipLaunchKernelGGL(attn_kernel,       dim3(256),  dim3(256), 0, stream,
                       adj, hT, f1, f2, mr, si, out);
}
